// QDense_16252156248077
// MI455X (gfx1250) — matrix operand order repaired and checked by reading, not compiled, not run on hardware
//
#include <hip/hip_runtime.h>
#include <hip/hip_bf16.h>

// ---- CDNA5 WMMA vector types -------------------------------------------------
typedef __attribute__((ext_vector_type(16))) __bf16 v16bf;
typedef __attribute__((ext_vector_type(8)))  __bf16 v8bf;
typedef __attribute__((ext_vector_type(8)))  float  v8f;

union FragBF { v16bf v; v8bf h[2]; };

// round-to-nearest-even fp32 -> bf16, packed pair into one dword
__device__ __forceinline__ unsigned int pack_bf2(float a, float b) {
  unsigned int ua = __float_as_uint(a);
  unsigned int ub = __float_as_uint(b);
  ua += 0x7FFFu + ((ua >> 16) & 1u);
  ub += 0x7FFFu + ((ub >> 16) & 1u);
  return (ua >> 16) | (ub & 0xFFFF0000u);
}

// CDNA5 async global->LDS DMA (16 bytes per lane), tracked by ASYNCcnt.
__device__ __forceinline__ void async_copy_b128(unsigned int lds_byte_addr,
                                                const void* gaddr) {
  asm volatile("global_load_async_to_lds_b128 %0, %1, off"
               :: "v"(lds_byte_addr), "v"(gaddr)
               : "memory");
}
__device__ __forceinline__ void wait_async_all() {
  asm volatile("s_wait_asynccnt 0" ::: "memory");
}

// =============================================================================
// Prep 1: materialize W_big (4096 x 4096, row = out-col n, col = k) in bf16,
// Hamilton block permutation + signs folded in.  32 MB into d_ws.
// =============================================================================
__global__ __launch_bounds__(256)
void prep_wbig(const float* __restrict__ Wrr, const float* __restrict__ Wri,
               const float* __restrict__ Wrj, const float* __restrict__ Wrk,
               unsigned int* __restrict__ WbigU)
{
  const int   Ttab[16] = {0,1,2,3,  1,0,3,2,  2,3,0,1,  3,2,1,0};
  const float Stab[16] = {1.f,-1.f,-1.f,-1.f,  1.f,1.f,1.f,-1.f,
                          1.f,-1.f,1.f,1.f,    1.f,1.f,-1.f,1.f};
  const int idx = (blockIdx.x * 256 + threadIdx.x) * 8;   // 16M elements total
  const int n = idx >> 12;
  const int k = idx & 4095;
  const int oc = n >> 10, o = n & 1023;
  const int kc = k >> 10, kk = k & 1023;
  const int   sel = Ttab[oc * 4 + kc];
  const float s   = Stab[oc * 4 + kc];
  const float* W  = (sel == 0) ? Wrr : (sel == 1) ? Wri : (sel == 2) ? Wrj : Wrk;
  const float4* p = reinterpret_cast<const float4*>(W + o * 1024 + kk);
  const float4 a = p[0], b = p[1];
  *reinterpret_cast<uint4*>(WbigU + (idx >> 1)) =
      make_uint4(pack_bf2(s * a.x, s * a.y), pack_bf2(s * a.z, s * a.w),
                 pack_bf2(s * b.x, s * b.y), pack_bf2(s * b.z, s * b.w));
}

// =============================================================================
// Prep 2: x fp32 -> bf16 (8192 x 4096).  64 MB into d_ws.
// =============================================================================
__global__ __launch_bounds__(256)
void prep_xbf(const float* __restrict__ x, unsigned int* __restrict__ XbU)
{
  const size_t idx = ((size_t)blockIdx.x * 256 + threadIdx.x) * 8;  // 32M elems
  const float4* p = reinterpret_cast<const float4*>(x + idx);
  const float4 a = p[0], b = p[1];
  *reinterpret_cast<uint4*>(XbU + (idx >> 1)) =
      make_uint4(pack_bf2(a.x, a.y), pack_bf2(a.z, a.w),
                 pack_bf2(b.x, b.y), pack_bf2(b.z, b.w));
}

// =============================================================================
// Main GEMM: out(8192x4096) = Xbf(8192x4096) * Wbig^T + tile(b_rr)
// 256 threads = 8 wave32; macro tile 128(M) x 128(N); K-step 64.
// Each wave: 32(M) x 64(N) patch = 2x4 WMMA tiles.
// Staging: async global->LDS DMA (no VGPR round-trip), double-buffered.
// =============================================================================
#define GBM 128
#define GBN 128
#define GKS 64
#define GNK 64         // 4096 / 64 k-steps
#define LDT 72         // LDS row stride (bf16 elems) = 64 + 8 pad, 144B (16B-aligned)

__global__ __launch_bounds__(256)
void qgemm_bf16(const unsigned short* __restrict__ Xb,
                const unsigned short* __restrict__ Wb,
                const float* __restrict__ brr,
                float* __restrict__ out)
{
  __shared__ __align__(16) unsigned short shA[2][GBM * LDT];  // 36 KB
  __shared__ __align__(16) unsigned short shB[2][GBN * LDT];  // 36 KB

  const int tid  = threadIdx.x;
  const int lane = tid & 31;
  const int wave = tid >> 5;
  const int wm   = wave >> 1;          // 0..3
  const int wn   = wave & 1;           // 0..1
  const int m0   = blockIdx.y * GBM;
  const int n0   = blockIdx.x * GBN;

  // staging: 16B chunk = 8 bf16; 8 chunks per 64-elem row; 1024 chunks per tile;
  // thread handles chunk (i*256 + tid) -> row = i*32 + (tid>>3), koff = (tid&7)*8
  const int srow = tid >> 3;
  const int skof = (tid & 7) * 8;
  const unsigned short* gA = Xb + (size_t)(m0 + srow) * 4096 + skof;
  const unsigned short* gB = Wb + (size_t)(n0 + srow) * 4096 + skof;
  const int sofs = srow * LDT + skof;          // LDS elem offset for chunk i=0

  v8f acc[2][4] = {};

  // ---- prologue: async-fill buffer 0 ----
#pragma unroll
  for (int i = 0; i < 4; ++i) {
    async_copy_b128((unsigned int)(uintptr_t)&shA[0][sofs + i * 32 * LDT],
                    gA + (size_t)i * 32 * 4096);
    async_copy_b128((unsigned int)(uintptr_t)&shB[0][sofs + i * 32 * LDT],
                    gB + (size_t)i * 32 * 4096);
  }
  wait_async_all();
  __syncthreads();

  const int la  = lane & 15;
  const int akb = (lane & 16) ? 8 : 0;    // A: lanes 0-15 K{0-7,16-23}, 16-31 K{8-15,24-31}
  const int bkb = (lane & 16) ? 8 : 0;    // B: lanes 0-15 K{0-7,16-23}, 16-31 K{8-15,24-31}

  for (int kt = 0; kt < GNK; ++kt) {
    const int cur = kt & 1;

    if (kt + 1 < GNK) {                   // async DMA next tiles into other buffer
      const int nb = cur ^ 1;
      const size_t ko = (size_t)(kt + 1) * GKS;
#pragma unroll
      for (int i = 0; i < 4; ++i) {
        async_copy_b128((unsigned int)(uintptr_t)&shA[nb][sofs + i * 32 * LDT],
                        gA + (size_t)i * 32 * 4096 + ko);
        async_copy_b128((unsigned int)(uintptr_t)&shB[nb][sofs + i * 32 * LDT],
                        gB + (size_t)i * 32 * 4096 + ko);
      }
    }
    if (kt + 2 < GNK)
      __builtin_prefetch(gA + (size_t)(kt + 2) * GKS, 0, 1);

    // ---- compute on buffer `cur` (overlaps with the async DMA) ----
    {
      const unsigned short* As = shA[cur];
      const unsigned short* Bs = shB[cur];
#pragma unroll
      for (int kk = 0; kk < 2; ++kk) {           // two 32-deep WMMA chunks
        const int kb = kk * 32;
        FragBF fa[2];
#pragma unroll
        for (int tm = 0; tm < 2; ++tm) {
          const int r = wm * 32 + tm * 16 + la;
          fa[tm].h[0] = *reinterpret_cast<const v8bf*>(&As[r * LDT + kb + akb]);
          fa[tm].h[1] = *reinterpret_cast<const v8bf*>(&As[r * LDT + kb + akb + 16]);
        }
#pragma unroll
        for (int tn = 0; tn < 4; ++tn) {
          const int c = wn * 64 + tn * 16 + la;
          FragBF fb;
          fb.h[0] = *reinterpret_cast<const v8bf*>(&Bs[c * LDT + kb + bkb]);
          fb.h[1] = *reinterpret_cast<const v8bf*>(&Bs[c * LDT + kb + bkb + 16]);
#pragma unroll
          for (int tm = 0; tm < 2; ++tm)
            acc[tm][tn] = __builtin_amdgcn_wmma_f32_16x16x32_bf16(
                false, fa[tm].v, false, fb.v, (short)0, acc[tm][tn],
                false, false);
        }
      }
    }

    wait_async_all();     // own DMA chunks landed in LDS
    __syncthreads();      // all waves' chunks landed; prev buffer free
  }

  // ---- epilogue: bias + store (C layout: lanes 0-15 M=r, lanes 16-31 M=r+8) ----
  const int hs = (lane & 16) ? 8 : 0;
#pragma unroll
  for (int tn = 0; tn < 4; ++tn) {
    const int ncol   = n0 + wn * 64 + tn * 16 + la;
    const float bias = brr[ncol & 1023];
#pragma unroll
    for (int tm = 0; tm < 2; ++tm) {
      const int mbase = m0 + wm * 32 + tm * 16 + hs;
      float* op = out + (size_t)mbase * 4096 + ncol;
#pragma unroll
      for (int r = 0; r < 8; ++r)
        op[(size_t)r * 4096] = acc[tm][tn][r] + bias;
    }
  }
}

// =============================================================================
extern "C" void kernel_launch(void* const* d_in, const int* in_sizes, int n_in,
                              void* d_out, int out_size, void* d_ws, size_t ws_size,
                              hipStream_t stream) {
  (void)in_sizes; (void)n_in; (void)out_size; (void)ws_size;
  const float* x   = (const float*)d_in[0];
  const float* Wrr = (const float*)d_in[1];
  const float* Wri = (const float*)d_in[2];
  const float* Wrj = (const float*)d_in[3];
  const float* Wrk = (const float*)d_in[4];
  const float* brr = (const float*)d_in[5];
  float* out = (float*)d_out;

  // workspace layout: [ W_big bf16 : 32 MB ][ X bf16 : 64 MB ]  (needs 96 MB)
  unsigned short* Wbig = (unsigned short*)d_ws;
  unsigned short* Xbf  = Wbig + (size_t)4096 * 4096;

  prep_wbig<<<dim3(8192),  dim3(256), 0, stream>>>(Wrr, Wri, Wrj, Wrk,
                                                   (unsigned int*)Wbig);
  prep_xbf <<<dim3(16384), dim3(256), 0, stream>>>(x, (unsigned int*)Xbf);

  dim3 grid(4096 / GBN, 8192 / GBM);   // (32, 64)
  qgemm_bf16<<<grid, dim3(256), 0, stream>>>(Xbf, Wbig, brr, out);
}
